// ATAB_89859305767670
// MI455X (gfx1250) — matrix operand order repaired and checked by reading, not compiled, not run on hardware
//
#include <hip/hip_runtime.h>

typedef _Float16 v16h __attribute__((ext_vector_type(16)));
typedef _Float16 v8h  __attribute__((ext_vector_type(8)));
typedef float    v8f  __attribute__((ext_vector_type(8)));

#define B_  8
#define H_  128
#define W_  256
#define C_  64
#define F_  64

// LDS strides in halves; all multiples of 8 halves (16B) for b128-aligned frag loads,
// and not multiples of 64 banks to stagger per-lane rows.
#define XCOLS     264      // 256 + 2*2 halo (dilation 2) rounded up
#define XSTRIDE   72       // channel row stride (64 data + 8 pad)
#define WTSTRIDE  584      // 576 K + 8 pad
#define KQSTRIDE  72       // 64 channels + 8 pad
#define VSTRIDE   264      // 256 keys + 8 pad
#define SSTRIDE   257      // f32 scores row stride

// ---- WMMA fragment loaders (v_wmma_f32_16x16x32_f16 layouts, ISA 7.12.2) ----
// A (16x32 f16): lane<16 -> K = {half*8+0..7, 16+half*8+0..7} of its row.
static __device__ inline v16h fragA(const _Float16* row_k0, int half) {
  const _Float16* p = row_k0 + half * 8;
  v8h lo = *(const v8h*)(p);
  v8h hi = *(const v8h*)(p + 16);
  return __builtin_shufflevector(lo, hi, 0,1,2,3,4,5,6,7,8,9,10,11,12,13,14,15);
}
// B (32x16 f16): lane holds column n=lane&15; K = {half*8+0..7, 16+half*8+0..7}.
static __device__ inline v16h fragB(const _Float16* col_k0, int half) {
  const _Float16* p = col_k0 + half * 8;
  v8h lo = *(const v8h*)(p);
  v8h hi = *(const v8h*)(p + 16);
  return __builtin_shufflevector(lo, hi, 0,1,2,3,4,5,6,7,8,9,10,11,12,13,14,15);
}

// =====================================================================
// Kernel 1: dilated 3x3 conv (dil=2, SAME) as implicit GEMM on WMMA.
// grid.x = B*H (one (b,h) output row of 256 pixels), grid.y = 0/1/2 -> Q/K/V
// block = 256 threads (8 waves). Output f16 to workspace.
// =====================================================================
__global__ __launch_bounds__(256) void qkv_conv_wmma(
    const float* __restrict__ X,
    const float* __restrict__ Wq, const float* __restrict__ Wk, const float* __restrict__ Wv,
    const float* __restrict__ bq, const float* __restrict__ bk, const float* __restrict__ bv,
    _Float16* __restrict__ qo, _Float16* __restrict__ ko, _Float16* __restrict__ vo)
{
  extern __shared__ char smemraw[];
  _Float16* Xs = (_Float16*)smemraw;                 // [3][XCOLS][XSTRIDE]
  _Float16* WT = Xs + 3 * XCOLS * XSTRIDE;           // [64][WTSTRIDE]  (f-major, K minor)

  const int tid  = threadIdx.x;
  const int wave = tid >> 5;
  const int lane = tid & 31;
  const int lid  = lane & 15;
  const int half = lane >> 4;

  const int bh = blockIdx.x;
  const int b  = bh / H_;
  const int h  = bh - b * H_;
  const int which = blockIdx.y;

  const float* Wsel = (which == 0) ? Wq : (which == 1) ? Wk : Wv;
  const float* bsel = (which == 0) ? bq : (which == 1) ? bk : bv;
  _Float16*    osel = (which == 0) ? qo : (which == 1) ? ko : vo;

  // CDNA5 global prefetch of the next row of X (global_prefetch_b8 path).
  {
    int hn = (h + 1 < H_) ? h + 1 : h;
    __builtin_prefetch(X + (((size_t)b * H_ + hn) * W_) * C_ + tid * 64, 0, 1);
  }

  // Stage weights transposed: WT[f][k], k = (kh*3+kw)*64 + c. Global is HWIO f32.
  for (int idx = tid; idx < 9 * 64 * 64; idx += 256) {
    int f   = idx & 63;
    int c   = (idx >> 6) & 63;
    int kwh = idx >> 12;                       // 0..8
    WT[f * WTSTRIDE + kwh * 64 + c] = (_Float16)Wsel[idx];
  }
  // Stage 3 input rows (h-2, h, h+2) with +-2 column halo, zero-padded, f16.
  for (int idx = tid; idx < 3 * 260 * 64; idx += 256) {
    int c   = idx & 63;
    int col = (idx >> 6) % 260;
    int r   = (idx >> 6) / 260;
    int hin = h + (r - 1) * 2;
    int win = col - 2;
    float v = 0.0f;
    if ((unsigned)hin < (unsigned)H_ && (unsigned)win < (unsigned)W_)
      v = X[(((size_t)b * H_ + hin) * W_ + win) * C_ + c];
    Xs[(r * XCOLS + col) * XSTRIDE + c] = (_Float16)v;
  }
  __syncthreads();

  // 64 output tiles (16 M-tiles of pixels x 4 N-tiles of channels), 8 per wave.
  const size_t obase = (size_t)bh * W_ * F_;
  for (int s = 0; s < 8; ++s) {
    const int t = wave * 8 + s;
    const int i = t >> 2;      // pixel tile 0..15
    const int j = t & 3;       // channel tile 0..3
    v8f acc = {};
    const _Float16* brow = WT + (j * 16 + lid) * WTSTRIDE;
    #pragma unroll
    for (int kc = 0; kc < 18; ++kc) {          // K = 576 in chunks of 32
      const int kh  = kc / 6;
      const int rem = kc - kh * 6;
      const int kw  = rem >> 1;
      const int c0  = (rem & 1) * 32;
      const _Float16* arow =
          Xs + ((kh * XCOLS) + (i * 16 + lid) + kw * 2) * XSTRIDE + c0;
      v16h a  = fragA(arow, half);
      v16h bb = fragB(brow + kc * 32, half);
      acc = __builtin_amdgcn_wmma_f32_16x16x32_f16(
          false, a, false, bb, (short)0, acc, false, false);
    }
    const float bias = bsel[j * 16 + lid];
    #pragma unroll
    for (int r = 0; r < 8; ++r) {
      const int m = i * 16 + r + 8 * half;     // pixel within the row
      osel[obase + (size_t)m * F_ + (j * 16 + lid)] = (_Float16)(acc[r] + bias);
    }
  }
}

// =====================================================================
// Kernel 2: fused row attention. grid.x = B*H*4 (64-query blocks),
// block = 256 threads (8 waves). scores (WMMA f16->f32) -> softmax (f32,
// quad-shuffle reductions) -> attn(f16) @ V (WMMA) -> f32 out.
// =====================================================================
__global__ __launch_bounds__(256) void row_attention_wmma(
    const _Float16* __restrict__ qws, const _Float16* __restrict__ kws,
    const _Float16* __restrict__ vws, float* __restrict__ out)
{
  extern __shared__ char smemraw[];
  _Float16* Ks  = (_Float16*)smemraw;              // [256][KQSTRIDE]
  _Float16* Qs  = Ks + 256 * KQSTRIDE;             // [64][KQSTRIDE]
  _Float16* Vt  = Qs + 64 * KQSTRIDE;              // [64][VSTRIDE]  (channel-major)
  float*    S   = (float*)(Vt + 64 * VSTRIDE);     // [64][SSTRIDE]
  _Float16* A16 = (_Float16*)(S + 64 * SSTRIDE);   // [64][VSTRIDE]

  const int tid  = threadIdx.x;
  const int wave = tid >> 5;
  const int lane = tid & 31;
  const int lid  = lane & 15;
  const int half = lane >> 4;

  const int q0 = (blockIdx.x & 3) * 64;            // query block within the row
  const int bh = blockIdx.x >> 2;                  // (b,h)
  const size_t rowbase = (size_t)bh * W_ * F_;

  // Stage K [key][ch] and V transposed [ch][key].
  for (int idx = tid; idx < 256 * 64; idx += 256) {
    int key = idx >> 6, c = idx & 63;
    Ks[key * KQSTRIDE + c] = kws[rowbase + (size_t)key * F_ + c];
    Vt[c * VSTRIDE + key]  = vws[rowbase + (size_t)key * F_ + c];
  }
  // Stage Q block [query][ch].
  for (int idx = tid; idx < 64 * 64; idx += 256) {
    int qq = idx >> 6, c = idx & 63;
    Qs[qq * KQSTRIDE + c] = qws[rowbase + (size_t)(q0 + qq) * F_ + c];
  }
  __syncthreads();

  // scores = Q @ K^T : 4x16 = 64 tiles of 16x16, K=64 (2 WMMA each); 8 per wave.
  for (int s = 0; s < 8; ++s) {
    const int t = wave * 8 + s;
    const int i = t >> 4;      // query tile 0..3
    const int j = t & 15;      // key tile 0..15
    v8f acc = {};
    const _Float16* arow = Qs + (i * 16 + lid) * KQSTRIDE;
    const _Float16* brow = Ks + (j * 16 + lid) * KQSTRIDE;
    #pragma unroll
    for (int kc = 0; kc < 2; ++kc) {
      v16h a  = fragA(arow + kc * 32, half);
      v16h bb = fragB(brow + kc * 32, half);
      acc = __builtin_amdgcn_wmma_f32_16x16x32_f16(
          false, a, false, bb, (short)0, acc, false, false);
    }
    #pragma unroll
    for (int r = 0; r < 8; ++r)
      S[(i * 16 + r + 8 * half) * SSTRIDE + (j * 16 + lid)] = acc[r];
  }
  __syncthreads();

  // softmax over 256 keys per row; 64 rows x 4 threads, quad shuffles.
  {
    const int row = tid >> 2;
    const int l4  = tid & 3;
    float* srow = S + row * SSTRIDE;
    float m = -3.402823466e38f;
    for (int jj = l4; jj < 256; jj += 4) m = fmaxf(m, srow[jj]);
    m = fmaxf(m, __shfl_xor(m, 1));
    m = fmaxf(m, __shfl_xor(m, 2));
    float ssum = 0.0f;
    for (int jj = l4; jj < 256; jj += 4) {
      float e = __expf(srow[jj] - m);
      srow[jj] = e;
      ssum += e;
    }
    ssum += __shfl_xor(ssum, 1);
    ssum += __shfl_xor(ssum, 2);
    const float inv = 1.0f / ssum;
    _Float16* arow = A16 + row * VSTRIDE;
    for (int jj = l4; jj < 256; jj += 4)
      arow[jj] = (_Float16)(srow[jj] * inv);
  }
  __syncthreads();

  // out = attn @ V : 4x4 = 16 tiles, K=256 (8 WMMA each); 2 per wave.
  const size_t obase = ((size_t)bh * W_ + q0) * F_;
  for (int s = 0; s < 2; ++s) {
    const int t = wave * 2 + s;
    const int i = t >> 2;      // query tile 0..3
    const int j = t & 3;       // channel tile 0..3
    v8f acc = {};
    const _Float16* arow = A16 + (i * 16 + lid) * VSTRIDE;
    const _Float16* brow = Vt  + (j * 16 + lid) * VSTRIDE;
    #pragma unroll
    for (int kk = 0; kk < 8; ++kk) {
      v16h a  = fragA(arow + kk * 32, half);
      v16h bb = fragB(brow + kk * 32, half);
      acc = __builtin_amdgcn_wmma_f32_16x16x32_f16(
          false, a, false, bb, (short)0, acc, false, false);
    }
    #pragma unroll
    for (int r = 0; r < 8; ++r) {
      const int m = i * 16 + r + 8 * half;
      out[obase + (size_t)m * F_ + (j * 16 + lid)] = acc[r];
    }
  }
}

// =====================================================================
// Host launcher
// =====================================================================
extern "C" void kernel_launch(void* const* d_in, const int* in_sizes, int n_in,
                              void* d_out, int out_size, void* d_ws, size_t ws_size,
                              hipStream_t stream) {
  (void)in_sizes; (void)n_in; (void)out_size; (void)ws_size;
  const float* X  = (const float*)d_in[0];
  const float* Wq = (const float*)d_in[1];
  const float* bq = (const float*)d_in[2];
  const float* Wk = (const float*)d_in[3];
  const float* bk = (const float*)d_in[4];
  const float* Wv = (const float*)d_in[5];
  const float* bv = (const float*)d_in[6];
  float* out = (float*)d_out;

  const size_t NQ = (size_t)B_ * H_ * W_ * F_;     // 16,777,216 elems per tensor
  _Float16* qws = (_Float16*)d_ws;                 // f16 Q/K/V staging: 3*32 MiB
  _Float16* kws = qws + NQ;
  _Float16* vws = kws + NQ;

  const size_t convLds = (size_t)(3 * XCOLS * XSTRIDE + 64 * WTSTRIDE) * sizeof(_Float16);
  const size_t attnLds = (size_t)(256 * KQSTRIDE + 64 * KQSTRIDE + 64 * VSTRIDE) * sizeof(_Float16)
                       + (size_t)(64 * SSTRIDE) * sizeof(float)
                       + (size_t)(64 * VSTRIDE) * sizeof(_Float16);

  dim3 gridc(B_ * H_, 3);                          // 1024 rows x {Q,K,V}
  qkv_conv_wmma<<<gridc, 256, convLds, stream>>>(
      X, Wq, Wk, Wv, bq, bk, bv, qws, kws, vws);

  row_attention_wmma<<<B_ * H_ * 4, 256, attnLds, stream>>>(qws, kws, vws, out);
}
